// WaveNet_74509092651424
// MI455X (gfx1250) — matrix operand order repaired and checked by reading, not compiled, not run on hardware
//
#include <hip/hip_runtime.h>
#include <hip/hip_bf16.h>

// ---------------- constants ----------------
#define WSEQ   1024
#define WPAD   2048                        // 512 zero halo + 1024 + 512 zero halo
#define HALO   512
#define CCH    64
#define NLAY   10
#define NHEAD  4
#define NB     8
#define SCALE_QK 0.35355339059327373f      // 1/sqrt(8)
#define BNINV    0.9999950000374997f       // 1/sqrt(1+1e-5)

typedef __attribute__((ext_vector_type(8)))  float        v8f;
typedef __attribute__((ext_vector_type(2)))  float        v2f;
typedef __attribute__((ext_vector_type(4)))  unsigned int v4u;
typedef __attribute__((ext_vector_type(16))) __bf16       v16bf;

union BFrag { v16bf v; unsigned short u[16]; unsigned int w[8]; v4u q[2]; };

__device__ __forceinline__ unsigned short f2bf(float f) {
  union { float f; unsigned int i; } c; c.f = f;
  unsigned int r = c.i + 0x7FFFu + ((c.i >> 16) & 1u);   // RNE
  return (unsigned short)(r >> 16);
}

// ---------------- zero the halo columns of padded xT (deterministic per call) ----------------
__global__ void k_pad(unsigned short* __restrict__ xTp) {
  int id = blockIdx.x * blockDim.x + threadIdx.x;       // NB * 1024 pad cols * 64 ch
  if (id >= NB * 1024 * CCH) return;
  int c = id & 63, col = (id >> 6) & 1023, b = id >> 16;
  int padcol = (col < HALO) ? col : (col + WSEQ);       // [0,512) and [1536,2048)
  xTp[(size_t)(b * WPAD + padcol) * CCH + c] = 0;
}

// ---------------- cast input: x -> xf (f32 c-major) + xTp (bf16 w-major, padded) ----------------
__global__ void k_cast(const float* __restrict__ x, float* __restrict__ xf,
                       unsigned short* __restrict__ xTp, int n) {
  int i = blockIdx.x * blockDim.x + threadIdx.x;
  if (i >= n) return;
  float v = x[i];
  xf[i] = v;
  int b = i >> 16, c = (i >> 10) & 63, w = i & 1023;
  xTp[(size_t)(b * WPAD + HALO + w) * CCH + c] = f2bf(v);
}

// ---------------- pack conv weights into A-fragment order (bf16) ----------------
// layout: cwp[((lay*8 + mt)*6 + ks)*32 + lane][16]; K-dim order kk = t*64 + ic
__global__ void k_prep_w(const float* __restrict__ conv_w, unsigned short* __restrict__ cwp) {
  int id = blockIdx.x * blockDim.x + threadIdx.x;
  if (id >= NLAY * 8 * 6 * 32 * 16) return;
  int el = id & 15, lane = (id >> 4) & 31;
  int rest = id >> 9;
  int ks = rest % 6; rest /= 6;
  int mt = rest & 7, lay = rest >> 3;
  int v = el >> 1, e = el & 1, hsel = lane >> 4;
  int Ka = (hsel << 4) + el;
  int kk = ks * 32 + Ka;
  int t = kk >> 6, ic = kk & 63;
  int oc = mt * 16 + (lane & 15);
  cwp[id] = f2bf(conv_w[((lay * 128 + oc) * 64 + ic) * 3 + t]);
}

// ---------------- conv(128x192 GEMM) + BN + gate ----------------
// grid (WSEQ/16, NB), block 256 (8 waves). Wave m-tile = wave id.
__global__ void k_conv_gate(const unsigned short* __restrict__ xTp,
                            const unsigned short* __restrict__ cwp,
                            const float* __restrict__ conv_b,
                            const float* __restrict__ bn_g,
                            const float* __restrict__ bn_b,
                            float* __restrict__ gf, unsigned short* __restrict__ gT,
                            int lay, int dil) {
  __shared__ float ylds[128 * 16];
  const int b = blockIdx.y, w0 = blockIdx.x * 16;
  const int tid = threadIdx.x, wave = tid >> 5, lane = tid & 31, hsel = lane >> 4;
  v8f acc = {};
  for (int ks = 0; ks < 6; ++ks) {                      // K = 192 = 6 x 32
    BFrag A, B;
    // A: prepacked fragment, 2x b128
    const unsigned short* ap = cwp + ((((lay * 8 + wave) * 6 + ks) * 32 + lane) << 4);
    A.q[0] = *(const v4u*)(ap);
    A.q[1] = *(const v4u*)(ap + 8);
    // B: one contiguous 16-channel run; halo padding -> unconditional b128 loads
    int base = ks * 32 + (hsel << 4);                   // kk run base (mult of 16)
    int t = base >> 6, icb = base & 63;
    int w = HALO + w0 + (lane & 15) + (t - 1) * dil;    // always in [0, WPAD)
    const unsigned short* bp = xTp + (size_t)(b * WPAD + w) * CCH + icb;
    B.q[0] = *(const v4u*)(bp);
    B.q[1] = *(const v4u*)(bp + 8);
    acc = __builtin_amdgcn_wmma_f32_16x16x32_bf16(false, A.v, false, B.v,
                                                  (short)0, acc, false, false);
  }
  // epilogue: conv bias + BN affine -> LDS
#pragma unroll
  for (int r = 0; r < 8; ++r) {
    int oc = wave * 16 + r + (hsel << 3);
    float y = (acc[r] + conv_b[lay * 128 + oc]) * (bn_g[lay * 128 + oc] * BNINV)
              + bn_b[lay * 128 + oc];
    ylds[oc * 16 + (lane & 15)] = y;
  }
  __syncthreads();
  // gate: g = tanh(filt) * sigmoid(gate)
#pragma unroll
  for (int e = 0; e < 4; ++e) {
    int id = tid + 256 * e;                             // 64x16 outputs
    int c = id >> 4, col = id & 15;
    float f = ylds[c * 16 + col];
    float g = ylds[(c + 64) * 16 + col];
    float o = tanhf(f) * (1.f / (1.f + expf(-g)));
    int w = w0 + col;
    gf[(b * CCH + c) * WSEQ + w] = o;                   // residual (f32, c-major)
    gT[(b * WSEQ + w) * CCH + c] = f2bf(o);             // GEMM input (bf16, w-major)
  }
}

// ---------------- fused q/k/v projection: (320 x 64) x (64 x W) ----------------
// rows 0..31 = q(h,o), 32..63 = k(h,o), 64..319 = v(h,c)
__global__ void k_qkv(const unsigned short* __restrict__ gT,
                      const float* __restrict__ wq, const float* __restrict__ bq,
                      const float* __restrict__ wk, const float* __restrict__ bk,
                      const float* __restrict__ wv, const float* __restrict__ bv,
                      float* __restrict__ qkT, unsigned short* __restrict__ vb, int lay) {
  __shared__ v4u Alds[40 * 32 * 2];                     // fragment-ordered, 40 KB
  __shared__ float bias[320];
  const int b = blockIdx.y, w0 = blockIdx.x * 64;
  const int tid = threadIdx.x, wave = tid >> 5, lane = tid & 31, hsel = lane >> 4;
  unsigned short* As = (unsigned short*)Alds;
  // stage weights (f32 -> bf16) swizzled into A-fragment order
  for (int s = 0; s < 80; ++s) {
    int id = tid + 256 * s;                             // 20480 elements
    int el = id & 15, ln = (id >> 4) & 31, g = id >> 9; // g = mt*2+ks
    int ks = g & 1, mt = g >> 1;
    int v = el >> 1, e = el & 1, hs = ln >> 4;
    int Ka = (hs << 4) + el;
    int kk = ks * 32 + Ka;                              // channel c
    int row = mt * 16 + (ln & 15);
    float f;
    if (row < 32)      { int h = row >> 3, o = row & 7; f = wq[((lay * NHEAD + h) * 8 + o) * CCH + kk]; }
    else if (row < 64) { int r2 = row - 32; int h = r2 >> 3, o = r2 & 7; f = wk[((lay * NHEAD + h) * 8 + o) * CCH + kk]; }
    else               { int r3 = row - 64; int h = r3 >> 6, co = r3 & 63; f = wv[((lay * NHEAD + h) * 64 + co) * CCH + kk]; }
    As[id] = f2bf(f);
  }
  if (tid < 320) {
    int r = tid; float f;
    if (r < 32)      { int h = r >> 3, o = r & 7;  f = bq[(lay * NHEAD + h) * 8 + o]; }
    else if (r < 64) { int r2 = r - 32; int h = r2 >> 3, o = r2 & 7; f = bk[(lay * NHEAD + h) * 8 + o]; }
    else             { int r3 = r - 64; int h = r3 >> 6, co = r3 & 63; f = bv[(lay * NHEAD + h) * 64 + co]; }
    bias[r] = f;
  }
  __syncthreads();
  for (int s = 0; s < 10; ++s) {                        // 20 Mtiles x 4 Ntiles / 8 waves
    int t = wave * 10 + s;
    int mt = t % 20, nt = t / 20;
    v8f acc = {};
    for (int ks = 0; ks < 2; ++ks) {                    // K = 64
      BFrag A, B;
      int gidx = ((mt * 2 + ks) * 32 + lane) * 2;
      A.q[0] = Alds[gidx];
      A.q[1] = Alds[gidx + 1];
      int col = w0 + nt * 16 + (lane & 15);
      const unsigned short* bp = gT + (size_t)(b * WSEQ + col) * CCH + ks * 32 + (hsel << 4);
      B.q[0] = *(const v4u*)(bp);
      B.q[1] = *(const v4u*)(bp + 8);
      acc = __builtin_amdgcn_wmma_f32_16x16x32_bf16(false, A.v, false, B.v,
                                                    (short)0, acc, false, false);
    }
#pragma unroll
    for (int r = 0; r < 8; ++r) {
      int row = mt * 16 + r + (hsel << 3);
      int col = w0 + nt * 16 + (lane & 15);
      float val = acc[r] + bias[row];
      if (row < 64) qkT[(size_t)(b * WSEQ + col) * 64 + row] = val;    // q,k f32 w-major
      else          vb[(size_t)(b * 256 + row - 64) * WSEQ + col] = f2bf(val); // v bf16 c-major
    }
  }
}

// ---------------- softmax stats (row max + 1/sum) via f32 WMMA ----------------
// grid (W/32, NHEAD, NB); dynamic LDS: 32 x 1024 f32 = 128 KB (WGP has 320 KB)
__global__ void k_stats(const float* __restrict__ qkT, float* __restrict__ st) {
  extern __shared__ float S[];
  const int b = blockIdx.z, h = blockIdx.y, i0 = blockIdx.x * 32;
  const int tid = threadIdx.x, wave = tid >> 5, lane = tid & 31, hsel = lane >> 4;
  const int c0 = wave * 128;
  for (int mt = 0; mt < 2; ++mt)
    for (int nt = 0; nt < 8; ++nt) {
      v8f acc = {};
      for (int ks = 0; ks < 2; ++ks) {                  // K = 8 = 2 x 4
        int od = h * 8 + ks * 4 + (hsel << 1);
        v2f A = *(const v2f*)(qkT + (size_t)(b * WSEQ + i0 + mt * 16 + (lane & 15)) * 64 + od);
        v2f B = *(const v2f*)(qkT + (size_t)(b * WSEQ + c0 + nt * 16 + (lane & 15)) * 64 + 32 + od);
        acc = __builtin_amdgcn_wmma_f32_16x16x4_f32(false, A, false, B,
                                                    (short)0, acc, false, false);
      }
#pragma unroll
      for (int r = 0; r < 8; ++r) {
        int il = mt * 16 + r + (hsel << 3);
        S[il * WSEQ + c0 + nt * 16 + (lane & 15)] = acc[r] * SCALE_QK;
      }
    }
  __syncthreads();
  int row = tid >> 3, part = tid & 7, base = part * 128;
  float m = -3.4e38f;
  for (int j = 0; j < 128; ++j) m = fmaxf(m, S[row * WSEQ + base + j]);
  for (int off = 4; off >= 1; off >>= 1) m = fmaxf(m, __shfl_xor(m, off, 8));
  float z = 0.f;
  for (int j = 0; j < 128; ++j) z += expf(S[row * WSEQ + base + j] - m);
  for (int off = 4; off >= 1; off >>= 1) z += __shfl_xor(z, off, 8);
  if (part == 0) {
    int i = i0 + row;
    st[(((b * NHEAD) + h) * WSEQ + i) * 2 + 0] = m;
    st[(((b * NHEAD) + h) * WSEQ + i) * 2 + 1] = 1.f / z;
  }
}

// ---------------- attention output + residual ----------------
// grid (W/64, NB). attn tile staged in LDS in B-fragment order (ds_load_b128).
__global__ void k_attn(const float* __restrict__ qkT, const float* __restrict__ st,
                       const unsigned short* __restrict__ vb,
                       const float* __restrict__ gf,
                       float* __restrict__ xf, unsigned short* __restrict__ xTp) {
  __shared__ v4u attn4[4 * 32 * 2];                     // 4 regions x 32 lanes x 32B
  unsigned short* attns = (unsigned short*)attn4;
  const int b = blockIdx.y, j0 = blockIdx.x * 64;
  const int tid = threadIdx.x, wave = tid >> 5, lane = tid & 31, hsel = lane >> 4;
  v8f acc0 = {}, acc1 = {};
  const int mt0 = wave >> 2, nt0 = wave & 3;            // output tiles of 4x4
  const int mt1 = (wave + 8) >> 2, nt1 = (wave + 8) & 3;
  const int smt = wave >> 2, snt = wave & 3;            // this wave's score tile
  for (int h = 0; h < NHEAD; ++h) {
    const float* sb = st + (size_t)((b * NHEAD) + h) * WSEQ * 2;
    const unsigned short* vbb = vb + (size_t)(b * 256 + h * 64) * WSEQ;
    for (int I = 0; I < 32; ++I) {
      const int I0 = I * 32;
      // ---- scores (f32 WMMA, K=8) -> attn bf16 (swizzled) in LDS ----
      v8f sc = {};
      for (int ks = 0; ks < 2; ++ks) {
        int od = h * 8 + ks * 4 + (hsel << 1);
        v2f A = *(const v2f*)(qkT + (size_t)(b * WSEQ + I0 + smt * 16 + (lane & 15)) * 64 + od);
        v2f B = *(const v2f*)(qkT + (size_t)(b * WSEQ + j0 + snt * 16 + (lane & 15)) * 64 + 32 + od);
        sc = __builtin_amdgcn_wmma_f32_16x16x4_f32(false, A, false, B,
                                                   (short)0, sc, false, false);
      }
#pragma unroll
      for (int r = 0; r < 8; ++r) {
        int il = smt * 16 + r + (hsel << 3);
        v2f mz = *(const v2f*)(sb + (I0 + il) * 2);     // (m, 1/Z)
        float a = expf(sc[r] * SCALE_QK - mz[0]) * mz[1];
        // swizzled write: region snt, lane' = (j&15)|((i&8)<<1), elem = (i&7)|((i&16)>>1)
        attns[snt * 512 + (((lane & 15) | ((il & 8) << 1)) << 4) + ((il & 7) | ((il & 16) >> 1))] = f2bf(a);
      }
      __syncthreads();
      // prefetch next K-block of v for this wave's tiles
      __builtin_prefetch(vbb + (size_t)(mt0 * 16 + (lane & 15)) * WSEQ + I0 + 32, 0, 1);
      // ---- head_out += v[:,I-block] @ attn  (bf16 WMMA) ----
#pragma unroll
      for (int tt = 0; tt < 2; ++tt) {
        int mt = tt ? mt1 : mt0, nt = tt ? nt1 : nt0;
        BFrag A, B;
        const unsigned short* ap = vbb + (size_t)(mt * 16 + (lane & 15)) * WSEQ + I0 + (hsel << 3);
        A.q[0] = *(const v4u*)(ap);                     // K = hsel*8 + 0..7
        A.q[1] = *(const v4u*)(ap + 16);                // K = 16 + hsel*8 + 0..7
        B.q[0] = attn4[(nt * 32 + lane) * 2];
        B.q[1] = attn4[(nt * 32 + lane) * 2 + 1];
        if (tt == 0)
          acc0 = __builtin_amdgcn_wmma_f32_16x16x32_bf16(false, A.v, false, B.v,
                                                         (short)0, acc0, false, false);
        else
          acc1 = __builtin_amdgcn_wmma_f32_16x16x32_bf16(false, A.v, false, B.v,
                                                         (short)0, acc1, false, false);
      }
      __syncthreads();                                  // before attn tile reuse
    }
  }
  // epilogue: x = 2*g + sum_h head_out ; write xf (f32 c-major) + xTp (bf16 w-major, packed)
#pragma unroll
  for (int tt = 0; tt < 2; ++tt) {
    int mt = tt ? mt1 : mt0, nt = tt ? nt1 : nt0;
    v8f a = tt ? acc1 : acc0;
    int j = j0 + nt * 16 + (lane & 15);
    unsigned short pk[8];
#pragma unroll
    for (int r = 0; r < 8; ++r) {
      int c = mt * 16 + (hsel << 3) + r;
      float val = 2.f * gf[(b * CCH + c) * WSEQ + j] + a[r];
      xf[(b * CCH + c) * WSEQ + j] = val;
      pk[r] = f2bf(val);
    }
    *(v4u*)(xTp + (size_t)(b * WPAD + HALO + j) * CCH + mt * 16 + (hsel << 3)) = *(v4u*)pk;
  }
}

// ---------------- final 1xC projection ----------------
__global__ void k_final(const float* __restrict__ xf, const float* __restrict__ ow,
                        const float* __restrict__ ob, float* __restrict__ out) {
  int id = blockIdx.x * blockDim.x + threadIdx.x;
  if (id >= NB * WSEQ) return;
  int b = id >> 10, w = id & 1023;
  float s = ob[0];
  for (int c = 0; c < CCH; ++c) s += ow[c] * xf[(b * CCH + c) * WSEQ + w];
  out[id] = s;
}

// ---------------- host ----------------
extern "C" void kernel_launch(void* const* d_in, const int* in_sizes, int n_in,
                              void* d_out, int out_size, void* d_ws, size_t ws_size,
                              hipStream_t stream) {
  const float* x      = (const float*)d_in[0];
  const float* conv_w = (const float*)d_in[1];
  const float* conv_b = (const float*)d_in[2];
  const float* bn_g   = (const float*)d_in[3];
  const float* bn_b   = (const float*)d_in[4];
  const float* wq     = (const float*)d_in[5];
  const float* bq     = (const float*)d_in[6];
  const float* wk     = (const float*)d_in[7];
  const float* bk     = (const float*)d_in[8];
  const float* wv     = (const float*)d_in[9];
  const float* bv     = (const float*)d_in[10];
  const float* out_w  = (const float*)d_in[11];
  const float* out_b  = (const float*)d_in[12];

  char* ws = (char*)d_ws;
  float*          xf  = (float*)         (ws + 0);            // 2 MB
  unsigned short* xTp = (unsigned short*)(ws + 2097152);      // 2 MB  (w-major, padded)
  float*          gf  = (float*)         (ws + 4194304);      // 2 MB
  unsigned short* gT  = (unsigned short*)(ws + 6291456);      // 1 MB  (w-major)
  float*          qkT = (float*)         (ws + 7340032);      // 2 MB  (w-major)
  unsigned short* vb  = (unsigned short*)(ws + 9437184);      // 4 MB  (c-major)
  float*          st  = (float*)         (ws + 13631488);     // 256 KB
  unsigned short* cwp = (unsigned short*)(ws + 13893632);     // 480 KB packed conv weights

  int n = NB * CCH * WSEQ;
  k_pad<<<dim3((NB * 1024 * CCH + 255) / 256), 256, 0, stream>>>(xTp);
  k_cast<<<dim3((n + 255) / 256), 256, 0, stream>>>(x, xf, xTp, n);
  k_prep_w<<<dim3((NLAY * 8 * 6 * 32 * 16 + 255) / 256), 256, 0, stream>>>(conv_w, cwp);

  for (int i = 0; i < NLAY; ++i) {
    int d = 1 << i;
    k_conv_gate<<<dim3(WSEQ / 16, NB), 256, 0, stream>>>(xTp, cwp, conv_b,
                                                         bn_g, bn_b, gf, gT, i, d);
    k_qkv<<<dim3(WSEQ / 64, NB), 256, 0, stream>>>(gT, wq, bq, wk, bk, wv, bv,
                                                   qkT, vb, i);
    k_stats<<<dim3(WSEQ / 32, NHEAD, NB), 256, 32 * WSEQ * sizeof(float), stream>>>(qkT, st);
    k_attn<<<dim3(WSEQ / 64, NB), 256, 0, stream>>>(qkT, st, vb, gf, xf, xTp);
  }
  k_final<<<dim3((NB * WSEQ + 255) / 256), 256, 0, stream>>>(xf, out_w, out_b,
                                                             (float*)d_out);
}
